// Attention_80212809220238
// MI455X (gfx1250) — matrix operand order repaired and checked by reading, not compiled, not run on hardware
//
#include <hip/hip_runtime.h>
#include <hip/hip_bf16.h>

typedef _Float16 half8  __attribute__((ext_vector_type(8)));
typedef _Float16 half16 __attribute__((ext_vector_type(16)));
typedef float    v8f    __attribute__((ext_vector_type(8)));

#define DIM 768
#define HEADS 12
#define HEAD_DIM 64
#define BATCH 8
#define SEQ 1024
#define TOKENS (BATCH * SEQ)        // 8192
#define C3 (3 * DIM)                // 2304

static __device__ __forceinline__ half16 cat8(half8 a, half8 b) {
  return __builtin_shufflevector(a, b, 0,1,2,3,4,5,6,7,8,9,10,11,12,13,14,15);
}

static __device__ __forceinline__ v8f wmma16(half16 a, half16 b, v8f c) {
  // D = A(16x32,f16) x B(32x16,f16) + C(16x16,f32)
  return __builtin_amdgcn_wmma_f32_16x16x32_f16(
      /*neg_a=*/false, a, /*neg_b=*/false, b,
      /*c_mod=*/(short)0, c, /*reuse_a=*/false, /*reuse_b=*/false);
}

// Async global->LDS copy of 32 bytes (two b128, same immediate offsets apply
// to both the LDS and global address per CDNA5 ISA 10.7/15.18.3).
static __device__ __forceinline__ void async_copy_b32x1(unsigned lds_off,
                                                        const void* gaddr) {
  asm volatile("global_load_async_to_lds_b128 %0, %1, off\n\t"
               "global_load_async_to_lds_b128 %0, %1, off offset:16"
               :: "v"(lds_off), "v"(gaddr) : "memory");
}

static __device__ __forceinline__ void async_wait0() {
  asm volatile("s_wait_asynccnt 0" ::: "memory");
}

static __device__ __forceinline__ unsigned lds_off_of(const void* p) {
  // Generic pointers to LDS carry the offset in the low 32 bits (flat
  // aperture mapping discards the upper bits).
  return (unsigned)(uintptr_t)p;
}

// ---------------------------------------------------------------- convert
__global__ void cvt_f16_kernel(const float* __restrict__ s,
                               _Float16* __restrict__ d, int n) {
  int i = blockIdx.x * blockDim.x + threadIdx.x;
  if (i < n) d[i] = (_Float16)s[i];
}

// ---------------------------------------------------------------- GEMM (A[M,K] * B[N,K]^T + bias)
// One wave computes one 16x64 output tile: A fragment loaded once per K-step
// and reused against 4 B fragments (4 WMMAs per A load). 4 waves / block.
template <typename OutT>
__global__ __launch_bounds__(128) void gemm_nt_kernel(
    const _Float16* __restrict__ A, const _Float16* __restrict__ B,
    const float* __restrict__ bias, OutT* __restrict__ C,
    int M, int N, int K) {
  const int lane = threadIdx.x & 31;
  const int wv   = threadIdx.x >> 5;
  const int tiles_n = N >> 6;                 // 64-wide super-tiles
  const int tile = blockIdx.x * 4 + wv;
  const int tm = tile / tiles_n;
  const int tn = tile % tiles_n;
  const int ln15 = lane & 15;
  const int hi   = lane >> 4;
  const int lo   = hi * 8;    // unused; A-frag also uses ko
  const int ko   = hi * 16;   // B-frag K offset

  const _Float16* arow = A + (size_t)(tm * 16 + ln15) * K;
  const _Float16* brow = B + (size_t)(tn * 64 + ln15) * K;
  const size_t bstep = (size_t)16 * K;        // 16 rows of B per sub-tile

  v8f c0 = {}, c1 = {}, c2 = {}, c3 = {};
  for (int k0 = 0; k0 < K; k0 += 32) {
    half16 a = cat8(*(const half8*)(arow + k0 + ko),
                    *(const half8*)(arow + k0 + ko + 8));
    const _Float16* bp = brow + k0 + ko;
    c0 = wmma16(a, cat8(*(const half8*)(bp),             *(const half8*)(bp + 8)),             c0);
    c1 = wmma16(a, cat8(*(const half8*)(bp + bstep),     *(const half8*)(bp + bstep + 8)),     c1);
    c2 = wmma16(a, cat8(*(const half8*)(bp + 2 * bstep), *(const half8*)(bp + 2 * bstep + 8)), c2);
    c3 = wmma16(a, cat8(*(const half8*)(bp + 3 * bstep), *(const half8*)(bp + 3 * bstep + 8)), c3);
  }

  v8f cc[4] = {c0, c1, c2, c3};
#pragma unroll
  for (int j = 0; j < 4; ++j) {
    const int   gcol = tn * 64 + j * 16 + ln15;
    const float bv   = bias[gcol];
#pragma unroll
    for (int r = 0; r < 8; ++r) {
      int grow = tm * 16 + r + hi * 8;
      C[(size_t)grow * N + gcol] = (OutT)(cc[j][r] + bv);
    }
  }
}

// ---------------------------------------------------------------- LayerNorm q/k + repack (v transposed)
static __device__ __forceinline__ float block64_sum(float v, float* red, int tid) {
  red[tid] = v;
  __syncthreads();
#pragma unroll
  for (int s = 32; s > 0; s >>= 1) {
    if (tid < s) red[tid] += red[tid + s];
    __syncthreads();
  }
  float r = red[0];
  __syncthreads();
  return r;
}

__global__ __launch_bounds__(64) void ln_pack_kernel(
    const _Float16* __restrict__ qkv,          // [TOKENS][3C] f16
    const float* __restrict__ qn_w, const float* __restrict__ qn_b,
    const float* __restrict__ kn_w, const float* __restrict__ kn_b,
    _Float16* __restrict__ q_h,                // [B*H][SEQ][64]
    _Float16* __restrict__ k_h,                // [B*H][SEQ][64]
    _Float16* __restrict__ v_t) {              // [B*H][64][SEQ]  (transposed)
  const int idx = blockIdx.x;                  // (b*SEQ+n)*HEADS + h
  const int h   = idx % HEADS;
  const int tok = idx / HEADS;
  const int b   = tok >> 10;
  const int n   = tok & (SEQ - 1);
  const int d   = threadIdx.x;

  const _Float16* row = qkv + (size_t)tok * C3;
  float qv = (float)row[h * HEAD_DIM + d];
  float kv = (float)row[DIM + h * HEAD_DIM + d];
  float vv = (float)row[2 * DIM + h * HEAD_DIM + d];

  __shared__ float red[64];
  const float inv64 = 1.0f / 64.0f;

  float mu_q  = block64_sum(qv, red, d) * inv64;
  float dq    = qv - mu_q;
  float var_q = block64_sum(dq * dq, red, d) * inv64;
  float qn    = dq * rsqrtf(var_q + 1e-5f) * qn_w[d] + qn_b[d];

  float mu_k  = block64_sum(kv, red, d) * inv64;
  float dk    = kv - mu_k;
  float var_k = block64_sum(dk * dk, red, d) * inv64;
  float kn    = dk * rsqrtf(var_k + 1e-5f) * kn_w[d] + kn_b[d];

  const int bh = b * HEADS + h;
  q_h[((size_t)bh * SEQ + n) * HEAD_DIM + d] = (_Float16)qn;
  k_h[((size_t)bh * SEQ + n) * HEAD_DIM + d] = (_Float16)kn;
  v_t[((size_t)bh * HEAD_DIM + d) * SEQ + n] = (_Float16)vv;
}

// ---------------------------------------------------------------- flash attention (per (b,h))
// 4 waves / block, one 16-query tile each. The block cooperatively stages
// each 32-key K tile (32x64) and V^T tile (64x32) into LDS with
// GLOBAL_LOAD_ASYNC_TO_LDS_B128; waves then build WMMA fragments from LDS.
__global__ __launch_bounds__(128) void attn_kernel(
    const _Float16* __restrict__ q_h, const _Float16* __restrict__ k_h,
    const _Float16* __restrict__ v_t, _Float16* __restrict__ attn_h) {
  const int tid   = threadIdx.x;
  const int lane  = tid & 31;
  const int wv    = tid >> 5;
  const int group = blockIdx.x & 15;         // 16 groups of 4 q-tiles per bh
  const int bh    = blockIdx.x >> 4;
  const int b     = bh / HEADS;
  const int h     = bh % HEADS;
  const int qt    = group * 4 + wv;          // query tile 0..63
  const int ln15  = lane & 15;
  const int hi    = lane >> 4;
  const int lo    = hi * 8;
  const int ko    = hi * 16;

  const _Float16* qb = q_h + (size_t)bh * SEQ * HEAD_DIM;
  const _Float16* kb = k_h + (size_t)bh * SEQ * HEAD_DIM;
  const _Float16* vb = v_t + (size_t)bh * HEAD_DIM * SEQ;

  __shared__ struct {
    alignas(16) _Float16 kt[32 * 64];        // K tile, row-major [key][d]
    alignas(16) _Float16 vt[64 * 32];        // V^T tile, row-major [d][key]
    alignas(16) _Float16 p[4][16 * 32];      // per-wave P staging
  } sm;
  _Float16* pbuf = sm.p[wv];

  // Per-thread async-copy source/dest decomposition (32 B per copy op).
  const int krow = tid >> 2, kseg = tid & 3;         // K: 32 rows x 4 segs
  const int vrow = tid >> 1, vseg = tid & 1;         // V: 64 rows x 2 segs
  const unsigned ldsK = lds_off_of(sm.kt + krow * 64 + kseg * 16);
  const unsigned ldsV = lds_off_of(sm.vt + vrow * 32 + vseg * 16);
  const _Float16* gK0 = kb + (size_t)krow * HEAD_DIM + kseg * 16;
  const _Float16* gV0 = vb + (size_t)vrow * SEQ + vseg * 16;

  // Load the two q A-fragments (K = 0..31 and 32..63) once.
  const _Float16* qrow = qb + (size_t)(qt * 16 + ln15) * HEAD_DIM;
  half16 aq0 = cat8(*(const half8*)(qrow + ko),      *(const half8*)(qrow + ko + 8));
  half16 aq1 = cat8(*(const half8*)(qrow + 32 + ko), *(const half8*)(qrow + 32 + ko + 8));

  float m_r[8], l_r[8];
  v8f   acc[4] = {};
#pragma unroll
  for (int r = 0; r < 8; ++r) { m_r[r] = -__builtin_inff(); l_r[r] = 0.0f; }

  const float scale = 0.125f;  // 64^-0.5

  for (int jc = 0; jc < SEQ / 32; ++jc) {
    const int keyBase = jc * 32;

    // Stage K/V tiles for this key chunk into LDS (async, ASYNCcnt-tracked).
    async_copy_b32x1(ldsK, gK0 + (size_t)keyBase * HEAD_DIM);
    async_copy_b32x1(ldsV, gV0 + keyBase);
    async_wait0();
    __syncthreads();

    // S tiles from LDS: keys [0,16) and [16,32) of the chunk.
    v8f s0 = {}, s1 = {};
    {
      const _Float16* kr0 = sm.kt + ln15 * 64;
      s0 = wmma16(aq0, cat8(*(const half8*)(kr0 + ko),      *(const half8*)(kr0 + ko + 8)),      s0);
      s0 = wmma16(aq1, cat8(*(const half8*)(kr0 + 32 + ko), *(const half8*)(kr0 + 32 + ko + 8)), s0);
      const _Float16* kr1 = sm.kt + (16 + ln15) * 64;
      s1 = wmma16(aq0, cat8(*(const half8*)(kr1 + ko),      *(const half8*)(kr1 + ko + 8)),      s1);
      s1 = wmma16(aq1, cat8(*(const half8*)(kr1 + 32 + ko), *(const half8*)(kr1 + 32 + ko + 8)), s1);
    }

    // Online softmax update per row (rows live in 16-lane half-wave groups).
#pragma unroll
    for (int r = 0; r < 8; ++r) {
      float v0 = s0[r] * scale;
      float v1 = s1[r] * scale;
      float mx = fmaxf(v0, v1);
      mx = fmaxf(mx, __shfl_xor(mx, 1));
      mx = fmaxf(mx, __shfl_xor(mx, 2));
      mx = fmaxf(mx, __shfl_xor(mx, 4));
      mx = fmaxf(mx, __shfl_xor(mx, 8));
      float mn = fmaxf(m_r[r], mx);
      float sf = __expf(m_r[r] - mn);
      float p0 = __expf(v0 - mn);
      float p1 = __expf(v1 - mn);
      float rs = p0 + p1;
      rs += __shfl_xor(rs, 1);
      rs += __shfl_xor(rs, 2);
      rs += __shfl_xor(rs, 4);
      rs += __shfl_xor(rs, 8);
      l_r[r] = l_r[r] * sf + rs;
      m_r[r] = mn;
#pragma unroll
      for (int t = 0; t < 4; ++t) acc[t][r] *= sf;
      const int m = r + hi * 8;
      pbuf[m * 32 + ln15]      = (_Float16)p0;
      pbuf[m * 32 + 16 + ln15] = (_Float16)p1;
    }
    __syncthreads();

    // Re-fragment P from LDS as a 16x32 A operand.
    half16 ap = cat8(*(const half8*)(pbuf + ln15 * 32 + ko),
                     *(const half8*)(pbuf + ln15 * 32 + ko + 8));

    // O += P(16x32) x V(32x64); V^T tile in LDS gives contiguous B-fragments.
#pragma unroll
    for (int t = 0; t < 4; ++t) {
      const _Float16* vr = sm.vt + (t * 16 + ln15) * 32 + ko;
      acc[t] = wmma16(ap, cat8(*(const half8*)(vr), *(const half8*)(vr + 8)), acc[t]);
    }
    __syncthreads();   // protect kt/vt/p before next chunk's overwrite
  }

  // Normalize and scatter to [B][N][C] (f16) for the projection GEMM.
#pragma unroll
  for (int r = 0; r < 8; ++r) {
    float inv = 1.0f / l_r[r];
    int   tok = qt * 16 + r + hi * 8;
#pragma unroll
    for (int t = 0; t < 4; ++t) {
      int col = h * HEAD_DIM + t * 16 + ln15;
      attn_h[((size_t)(b * SEQ + tok)) * DIM + col] = (_Float16)(acc[t][r] * inv);
    }
  }
}

// ---------------------------------------------------------------- host
extern "C" void kernel_launch(void* const* d_in, const int* in_sizes, int n_in,
                              void* d_out, int out_size, void* d_ws, size_t ws_size,
                              hipStream_t stream) {
  const float* x      = (const float*)d_in[0];
  const float* qkv_w  = (const float*)d_in[1];
  const float* qkv_b  = (const float*)d_in[2];
  const float* qn_w   = (const float*)d_in[3];
  const float* qn_b   = (const float*)d_in[4];
  const float* kn_w   = (const float*)d_in[5];
  const float* kn_b   = (const float*)d_in[6];
  const float* proj_w = (const float*)d_in[7];
  const float* proj_b = (const float*)d_in[8];
  float* out = (float*)d_out;

  char* ws = (char*)d_ws;
  _Float16* x_h     = (_Float16*)ws; ws += (size_t)TOKENS * DIM * 2;
  _Float16* qkvw_h  = (_Float16*)ws; ws += (size_t)C3 * DIM * 2;
  _Float16* projw_h = (_Float16*)ws; ws += (size_t)DIM * DIM * 2;
  _Float16* qkv     = (_Float16*)ws; ws += (size_t)TOKENS * C3 * 2;
  _Float16* q_h     = (_Float16*)ws; ws += (size_t)TOKENS * DIM * 2;
  _Float16* k_h     = (_Float16*)ws; ws += (size_t)TOKENS * DIM * 2;
  _Float16* v_t     = (_Float16*)ws; ws += (size_t)TOKENS * DIM * 2;
  _Float16* attn_h  = (_Float16*)ws; ws += (size_t)TOKENS * DIM * 2;

  // 1) f32 -> f16 conversions
  {
    int n = TOKENS * DIM;
    cvt_f16_kernel<<<(n + 255) / 256, 256, 0, stream>>>(x, x_h, n);
    n = C3 * DIM;
    cvt_f16_kernel<<<(n + 255) / 256, 256, 0, stream>>>(qkv_w, qkvw_h, n);
    n = DIM * DIM;
    cvt_f16_kernel<<<(n + 255) / 256, 256, 0, stream>>>(proj_w, projw_h, n);
  }

  // 2) QKV GEMM: [8192,768] x [2304,768]^T -> [8192,2304] (f16)
  {
    int tiles = (TOKENS / 16) * (C3 / 64);   // 512*36 = 18432
    gemm_nt_kernel<_Float16><<<tiles / 4, 128, 0, stream>>>(
        x_h, qkvw_h, qkv_b, qkv, TOKENS, C3, DIM);
  }

  // 3) LayerNorm q/k + repack (v transposed)
  ln_pack_kernel<<<TOKENS * HEADS, 64, 0, stream>>>(
      qkv, qn_w, qn_b, kn_w, kn_b, q_h, k_h, v_t);

  // 4) Flash attention per (b,h)
  attn_kernel<<<BATCH * HEADS * 16, 128, 0, stream>>>(q_h, k_h, v_t, attn_h);

  // 5) Output projection: [8192,768] x [768,768]^T -> [8192,768] (f32)
  {
    int tiles = (TOKENS / 16) * (DIM / 64);  // 512*12 = 6144
    gemm_nt_kernel<float><<<tiles / 4, 128, 0, stream>>>(
        attn_h, projw_h, proj_b, out, TOKENS, DIM, DIM);
  }
}
